// SimpleSkip_13134009991452
// MI455X (gfx1250) — matrix operand order repaired and checked by reading, not compiled, not run on hardware
//
#include <hip/hip_runtime.h>
#include <hip/hip_bf16.h>
#include <math.h>

typedef __attribute__((ext_vector_type(16))) __bf16 v16bf;
typedef __attribute__((ext_vector_type(8)))  float  v8f;

#define NNODES 100000
#define NEDGES 1600000
#define NF 6
#define HE 64
#define DD 32

static __device__ __forceinline__ v8f wmma_bf16(v16bf a, v16bf b, v8f c) {
    // D = A(16x32 bf16) * B(32x16 bf16) + C(16x16 f32)
    return __builtin_amdgcn_wmma_f32_16x16x32_bf16(
        /*neg_a=*/false, a, /*neg_b=*/false, b,
        /*c_mod=*/(short)0, c, /*reuse_a=*/false, /*reuse_b=*/false);
}

// tanh(x) = 1 - 2/(e^{2x}+1); clamped so e^{2x} stays finite.
// v_exp_f32 + v_rcp_f32 + 4 VALU, branchless.
static __device__ __forceinline__ float fast_tanh(float x) {
    const float cx = fminf(fmaxf(x, -15.0f), 15.0f);
    const float e2 = __expf(2.0f * cx);
    return 1.0f - 2.0f * __builtin_amdgcn_rcpf(e2 + 1.0f);
}

// ---------------------------------------------------------------------------
// Embed MLP: h = tanh(tanh(x@We1+be1)@We2+be2)   [N,6] -> [N,64] -> [N,32]
// One wave per 16 nodes. Layer1 on VALU (K=6) from LDS-staged We1/be1,
// h0 staged bf16 in LDS, layer2 (K=64) = two chained WMMA bf16 per col tile.
// We2 is staged in LDS pre-swizzled into WMMA B-fragment order so each lane
// grabs its fragment as one aligned 32B vector (2x ds_load_b128).
// ---------------------------------------------------------------------------
__global__ void k_embed(const float* __restrict__ x,
                        const float* __restrict__ We1, const float* __restrict__ be1,
                        const float* __restrict__ We2, const float* __restrict__ be2,
                        float* __restrict__ h, int n) {
    __shared__ __bf16 sh[8][16][HE];                       // 16 KB h0 staging
    __shared__ __align__(32) __bf16 swz2[2][2][32][16];    //  4 KB We2, B-frag order
    __shared__ float  w1s[NF * HE];                        //  1.5 KB We1
    __shared__ float  b1s[HE];

    // stage We2 pre-swizzled: swz2[kHalf][nt][lane][e] = We2[K][col]
    //   K = kHalf*32 + (lane>>4)*16 + e ; col = (lane&15) + nt*16
    for (int i = threadIdx.x; i < 2048; i += 256) {
        const int kh = i >> 10;
        const int nt = (i >> 9) & 1;
        const int ln = (i >> 4) & 31;
        const int e  = i & 15;
        const int K  = kh * 32 + (ln >> 4) * 16 + e;
        const int col = (ln & 15) + nt * 16;
        swz2[kh][nt][ln][e] = (__bf16)We2[K * DD + col];
    }
    for (int i = threadIdx.x; i < NF * HE; i += 256) w1s[i] = We1[i];
    if (threadIdx.x < HE) b1s[threadIdx.x] = be1[threadIdx.x];
    __syncthreads();

    const int wave = threadIdx.x >> 5;
    const int lane = threadIdx.x & 31;
    const int base = blockIdx.x * 128 + wave * 16;
    const int m    = lane & 15;          // M row of tile
    const int hi   = lane >> 4;          // half-wave
    const int node = base + m;
    const int nc   = node < n ? node : n - 1;

    // ---- layer 1: each lane computes 32 of 64 hidden feats of its node ----
    float xv[NF];
    #pragma unroll
    for (int k = 0; k < NF; ++k) xv[k] = x[nc * NF + k];
    const int fbase = hi * 32;
    #pragma unroll 8
    for (int j = 0; j < 32; ++j) {
        const int f = fbase + j;
        float acc = b1s[f];
        #pragma unroll
        for (int k = 0; k < NF; ++k) acc += xv[k] * w1s[k * HE + f];
        sh[wave][m][f] = (__bf16)fast_tanh(acc);
    }
    __syncthreads();

    // ---- A fragments: 16x32 bf16, K in [0,32) and [32,64) ----
    // lane<16 covers K = 0..15 ; lane>=16 covers K = 16..31
    const int kb = hi * 16;
    v16bf a0, a1;
    #pragma unroll
    for (int e = 0; e < 16; ++e) {
        const int K = kb + e;
        a0[e] = sh[wave][m][K];
        a1[e] = sh[wave][m][32 + K];
    }

    const bool full = (base + 16 <= n);  // wave-uniform fast path
    #pragma unroll
    for (int nt = 0; nt < 2; ++nt) {
        const int col = m + nt * 16;
        const v16bf b0 = *(const v16bf*)swz2[0][nt][lane];
        const v16bf b1 = *(const v16bf*)swz2[1][nt][lane];
        v8f c = {};
        c = wmma_bf16(a0, b0, c);
        c = wmma_bf16(a1, b1, c);
        const float bias = be2[col];
        if (full) {
            #pragma unroll
            for (int v = 0; v < 8; ++v)
                h[(base + v + hi * 8) * DD + col] = fast_tanh(c[v] + bias);
        } else {
            #pragma unroll
            for (int v = 0; v < 8; ++v) {
                const int nd = base + v + hi * 8;
                if (nd < n) h[nd * DD + col] = fast_tanh(c[v] + bias);
            }
        }
    }
}

// ---------------------------------------------------------------------------
// Degree / normalization: deg = 1 (self loop) + in-degree; dinv = rsqrt(deg)
// ---------------------------------------------------------------------------
__global__ void k_fill(float* __restrict__ p, float v, int nn) {
    const int i = blockIdx.x * blockDim.x + threadIdx.x;
    if (i < nn) p[i] = v;
}
__global__ void k_deg_edges(const int* __restrict__ ei, float* __restrict__ deg) {
    const int e = blockIdx.x * blockDim.x + threadIdx.x;
    if (e < NEDGES) atomicAdd(&deg[ei[NEDGES + e]], 1.0f);   // dst row
}
__global__ void k_rsqrt(float* __restrict__ p, int nn) {
    const int i = blockIdx.x * blockDim.x + threadIdx.x;
    if (i < nn) p[i] = __builtin_amdgcn_rsqf(p[i]);          // deg >= 1 always
}

// ---------------------------------------------------------------------------
// GCN linear: hw = hin @ W (WMMA bf16, K=32) ; acc = hw * dinv^2 (self loop)
// W staged in LDS pre-swizzled to B-fragment order.
// ---------------------------------------------------------------------------
__global__ void k_gcn_linear(const float* __restrict__ hin, const float* __restrict__ W,
                             const float* __restrict__ dinv,
                             float* __restrict__ hw, float* __restrict__ acc, int n) {
    __shared__ __align__(32) __bf16 swzW[2][32][16];       // 2 KB, B-frag order
    for (int i = threadIdx.x; i < 1024; i += 256) {
        const int nt = i >> 9;
        const int ln = (i >> 4) & 31;
        const int e  = i & 15;
        const int K  = (ln >> 4) * 16 + e;
        const int col = (ln & 15) + nt * 16;
        swzW[nt][ln][e] = (__bf16)W[K * DD + col];
    }
    __syncthreads();

    const int wave = threadIdx.x >> 5;
    const int lane = threadIdx.x & 31;
    const int base = blockIdx.x * 128 + wave * 16;
    const int m    = lane & 15;
    const int hi   = lane >> 4;
    const int node = base + m;
    const int nc   = node < n ? node : n - 1;

    const int kb = hi * 16;
    const float* row = hin + (size_t)nc * DD;
    v16bf a;
    #pragma unroll
    for (int e = 0; e < 8; ++e) {
        a[e]     = (__bf16)row[kb + e];
        a[e + 8] = (__bf16)row[kb + 8 + e];
    }

    // self-loop coefficients, invariant across the two column tiles
    float di2[8];
    #pragma unroll
    for (int v = 0; v < 8; ++v) {
        const int nd = base + v + hi * 8;
        const float di = dinv[nd < n ? nd : n - 1];
        di2[v] = di * di;
    }

    const bool full = (base + 16 <= n);
    #pragma unroll
    for (int nt = 0; nt < 2; ++nt) {
        const int col = m + nt * 16;
        const v16bf b = *(const v16bf*)swzW[nt][lane];
        v8f c = {};
        c = wmma_bf16(a, b, c);
        if (full) {
            #pragma unroll
            for (int v = 0; v < 8; ++v) {
                const size_t o = (size_t)(base + v + hi * 8) * DD + col;
                const float val = c[v];
                hw[o]  = val;
                acc[o] = val * di2[v];
            }
        } else {
            #pragma unroll
            for (int v = 0; v < 8; ++v) {
                const int nd = base + v + hi * 8;
                if (nd < n) {
                    const size_t o = (size_t)nd * DD + col;
                    const float val = c[v];
                    hw[o]  = val;
                    acc[o] = val * di2[v];
                }
            }
        }
    }
}

// ---------------------------------------------------------------------------
// Edge scatter: acc[dst] += hw[src] * dinv[src]*dinv[dst].  8 threads/edge,
// float4 gather (L2-resident), 4 float atomics each; prefetch edge stream.
// ---------------------------------------------------------------------------
__global__ void k_edge_scatter(const int* __restrict__ ei, const float* __restrict__ hw,
                               const float* __restrict__ dinv, float* __restrict__ acc) {
    const long long tid = (long long)blockIdx.x * blockDim.x + threadIdx.x;
    if (tid >= (long long)NEDGES * 8) return;
    const int e = (int)(tid >> 3);
    const int f = ((int)tid & 7) * 4;
    __builtin_prefetch(ei + e + 4096, 0, 1);           // global_prefetch_b8
    const int s = ei[e];
    const int d = ei[NEDGES + e];
    const float norm = dinv[s] * dinv[d];
    const float4 v = *(const float4*)(hw + (size_t)s * DD + f);
    float* dst = acc + (size_t)d * DD + f;
    atomicAdd(dst + 0, v.x * norm);
    atomicAdd(dst + 1, v.y * norm);
    atomicAdd(dst + 2, v.z * norm);
    atomicAdd(dst + 3, v.w * norm);
}

__global__ void k_bias_relu(const float4* __restrict__ acc, const float* __restrict__ b,
                            float4* __restrict__ out, int total4) {
    const int i = blockIdx.x * blockDim.x + threadIdx.x;
    if (i < total4) {
        float4 v = acc[i];
        const float4 bb = *(const float4*)(b + ((i * 4) & (DD - 1)));
        v.x = fmaxf(v.x + bb.x, 0.0f);
        v.y = fmaxf(v.y + bb.y, 0.0f);
        v.z = fmaxf(v.z + bb.z, 0.0f);
        v.w = fmaxf(v.w + bb.w, 0.0f);
        out[i] = v;
    }
}

// ---------------------------------------------------------------------------
// Predict: p = tanh(h@Wp1+bp1) via WMMA; out = tanh(p . Wp2 + bp2) via
// half-wave shfl_xor reduction (wave32).
// ---------------------------------------------------------------------------
__global__ void k_predict(const float* __restrict__ hin, const float* __restrict__ Wp1,
                          const float* __restrict__ bp1, const float* __restrict__ Wp2,
                          const float* __restrict__ bp2, float* __restrict__ out, int n) {
    __shared__ __align__(32) __bf16 swzW[2][32][16];
    for (int i = threadIdx.x; i < 1024; i += 256) {
        const int nt = i >> 9;
        const int ln = (i >> 4) & 31;
        const int e  = i & 15;
        const int K  = (ln >> 4) * 16 + e;
        const int col = (ln & 15) + nt * 16;
        swzW[nt][ln][e] = (__bf16)Wp1[K * DD + col];
    }
    __syncthreads();

    const int wave = threadIdx.x >> 5;
    const int lane = threadIdx.x & 31;
    const int base = blockIdx.x * 128 + wave * 16;
    const int m    = lane & 15;
    const int hi   = lane >> 4;
    const int node = base + m;
    const int nc   = node < n ? node : n - 1;

    const int kb = hi * 16;
    const float* row = hin + (size_t)nc * DD;
    v16bf a;
    #pragma unroll
    for (int e = 0; e < 8; ++e) {
        a[e]     = (__bf16)row[kb + e];
        a[e + 8] = (__bf16)row[kb + 8 + e];
    }
    float partial[8];
    #pragma unroll
    for (int v = 0; v < 8; ++v) partial[v] = 0.0f;

    #pragma unroll
    for (int nt = 0; nt < 2; ++nt) {
        const int col = m + nt * 16;
        const v16bf b = *(const v16bf*)swzW[nt][lane];
        v8f c = {};
        c = wmma_bf16(a, b, c);
        const float w2 = Wp2[col];
        const float bb = bp1[col];
        #pragma unroll
        for (int v = 0; v < 8; ++v) partial[v] += fast_tanh(c[v] + bb) * w2;
    }
    // reduce feature dim across each 16-lane half (xor masks stay in-half)
    const float ob = bp2[0];
    #pragma unroll
    for (int v = 0; v < 8; ++v) {
        float s = partial[v];
        s += __shfl_xor(s, 1, 32);
        s += __shfl_xor(s, 2, 32);
        s += __shfl_xor(s, 4, 32);
        s += __shfl_xor(s, 8, 32);
        const int nd = base + v + hi * 8;
        if (m == 0 && nd < n) out[nd] = fast_tanh(s + ob);
    }
}

// ---------------------------------------------------------------------------
extern "C" void kernel_launch(void* const* d_in, const int* in_sizes, int n_in,
                              void* d_out, int out_size, void* d_ws, size_t ws_size,
                              hipStream_t stream) {
    const float* x   = (const float*)d_in[0];
    const int*   ei  = (const int*)  d_in[1];
    const float* We1 = (const float*)d_in[2];
    const float* be1 = (const float*)d_in[3];
    const float* We2 = (const float*)d_in[4];
    const float* be2 = (const float*)d_in[5];
    const float* Wg1 = (const float*)d_in[6];
    const float* bg1 = (const float*)d_in[7];
    const float* Wg2 = (const float*)d_in[8];
    const float* bg2 = (const float*)d_in[9];
    const float* Wp1 = (const float*)d_in[10];
    const float* bp1 = (const float*)d_in[11];
    const float* Wp2 = (const float*)d_in[12];
    const float* bp2 = (const float*)d_in[13];
    float* out = (float*)d_out;

    // workspace layout (all L2-resident: 3 * 12.8MB + 0.4MB)
    float* h    = (float*)d_ws;                       // [N,32]
    float* hw   = h   + (size_t)NNODES * DD;          // [N,32]
    float* acc  = hw  + (size_t)NNODES * DD;          // [N,32]
    float* dinv = acc + (size_t)NNODES * DD;          // [N]

    const dim3 blk(256);
    const int gridNodes = (NNODES + 127) / 128;       // wave = 16 nodes, 8 waves/block
    const int gridN1    = (NNODES + 255) / 256;
    const int gridE     = (NEDGES + 255) / 256;
    const int gridNF4   = (NNODES * DD / 4 + 255) / 256;
    const long long sw  = (long long)NEDGES * 8;
    const int gridScat  = (int)((sw + 255) / 256);

    // embed MLP
    k_embed<<<gridNodes, blk, 0, stream>>>(x, We1, be1, We2, be2, h, NNODES);

    // symmetric normalization coefficients
    k_fill<<<gridN1, blk, 0, stream>>>(dinv, 1.0f, NNODES);
    k_deg_edges<<<gridE, blk, 0, stream>>>(ei, dinv);
    k_rsqrt<<<gridN1, blk, 0, stream>>>(dinv, NNODES);

    // GCN conv 1
    k_gcn_linear<<<gridNodes, blk, 0, stream>>>(h, Wg1, dinv, hw, acc, NNODES);
    k_edge_scatter<<<gridScat, blk, 0, stream>>>(ei, hw, dinv, acc);
    k_bias_relu<<<gridNF4, blk, 0, stream>>>((const float4*)acc, bg1, (float4*)h, NNODES * DD / 4);

    // GCN conv 2
    k_gcn_linear<<<gridNodes, blk, 0, stream>>>(h, Wg2, dinv, hw, acc, NNODES);
    k_edge_scatter<<<gridScat, blk, 0, stream>>>(ei, hw, dinv, acc);
    k_bias_relu<<<gridNF4, blk, 0, stream>>>((const float4*)acc, bg2, (float4*)h, NNODES * DD / 4);

    // predict MLP
    k_predict<<<gridNodes, blk, 0, stream>>>(h, Wp1, bp1, Wp2, bp2, out, NNODES);
}
